// MPNN_62508954026212
// MI455X (gfx1250) — compile-verified, matrix operand order checked by reading, not run on hardware
//
#include <hip/hip_runtime.h>
#include <hip/hip_bf16.h>

// ---------------------------------------------------------------------------
// MPNN forward for MI455X (gfx1250, wave32, WMMA f32<-f16 16x16x32).
// All GEMMs use v_wmma_f32_16x16x32_f16 (f16 A/B, f32 accumulate).
// BatchNorm (training mode): stats pass + fused recompute/normalize pass.
// This revision adds global_prefetch_b8 for the next grid-stride tile of the
// streaming inputs (ea / edge_index / X) to hide HBM latency.
// ---------------------------------------------------------------------------

typedef __attribute__((ext_vector_type(16))) _Float16 v16h;
typedef __attribute__((ext_vector_type(8)))  float    v8f;

#define NN 50000           // nodes
#define NE 800000          // edges
#define EA_TOT (NE + NN)   // edges + self loops
#define LATD 32
#define EPSBN 1e-5f

// CDNA5 16-bit A-matrix VGPR layout: lane holds row M=lane%16;
// VGPR v packs K pair: K = (v<4 ? 2v : 2v+8) + (lane>=16 ? 8 : 0)
__device__ __forceinline__ int kmap(int v, int lane) {
  return ((v < 4) ? (2 * v) : (2 * v + 8)) + ((lane & 16) ? 8 : 0);
}

__device__ __forceinline__ v8f wmma16(v16h a, v16h b, v8f c) {
  return __builtin_amdgcn_wmma_f32_16x16x32_f16(false, a, false, b, (short)0, c,
                                                false, false);
}

// A fragment from contiguous rows (row-major X, stride DIN), zero-pad K>=DIN.
// Branchless: clamped always-in-bounds loads + value select.
template <int DIN>
__device__ __forceinline__ v16h a_frag_rows(const float* __restrict__ X,
                                            long row0, int kc, int lane) {
  const float* p = X + (row0 + (lane & 15)) * (long)DIN;
  v16h a;
#pragma unroll
  for (int v = 0; v < 8; ++v) {
    int k = kc * 32 + kmap(v, lane);
    if constexpr (DIN % 32 == 0) {
      float2 f = *(const float2*)(p + k);   // 8B aligned: k even, row 8B-mult
      a[2 * v]     = (_Float16)f.x;
      a[2 * v + 1] = (_Float16)f.y;
    } else {
      int k0 = (k < DIN) ? k : 0;
      int k1 = (k + 1 < DIN) ? (k + 1) : 0;
      float f0 = p[k0], f1 = p[k1];
      a[2 * v]     = (k < DIN)     ? (_Float16)f0 : (_Float16)0.0f;
      a[2 * v + 1] = (k + 1 < DIN) ? (_Float16)f1 : (_Float16)0.0f;
    }
  }
  return a;
}

// A fragment for message input: cat(h[dst], ea[e], h[src])  (96 features)
__device__ __forceinline__ v16h a_frag_msg(const float* __restrict__ h,
                                           const float* __restrict__ ea,
                                           int s, int d, long e, int kc,
                                           int lane) {
  const float* base = (kc == 0) ? (h + (long)d * LATD)
                    : (kc == 1) ? (ea + e * (long)LATD)
                                : (h + (long)s * LATD);
  v16h a;
#pragma unroll
  for (int v = 0; v < 8; ++v) {
    int k = kmap(v, lane);
    float2 f = *(const float2*)(base + k);
    a[2 * v]     = (_Float16)f.x;
    a[2 * v + 1] = (_Float16)f.y;
  }
  return a;
}

// A fragment from per-wave LDS staging tile [16][72] (already f16)
__device__ __forceinline__ v16h a_frag_stage(const _Float16* st, int kc,
                                             int lane) {
  const _Float16* p = st + (lane & 15) * 72;
  v16h a;
#pragma unroll
  for (int v = 0; v < 8; ++v) {
    int k = kc * 32 + kmap(v, lane);
    a[2 * v]     = p[k];
    a[2 * v + 1] = p[k + 1];
  }
  return a;
}

// B fragment from LDS weights (row-major din x dout, f16); lane holds col N.
// Branchless clamped LDS reads; guards fold away for full-size din/dout.
__device__ __forceinline__ v16h b_frag(const _Float16* Wl, int din, int dout,
                                       int kc, int nt, int lane) {
  int n = nt * 16 + (lane & 15);
  bool nok = (n < dout);
  int nc = nok ? n : 0;
  v16h b;
#pragma unroll
  for (int v = 0; v < 8; ++v) {
    int k = kc * 32 + kmap(v, lane);
    int k0 = (k < din) ? k : 0;
    int k1 = (k + 1 < din) ? (k + 1) : 0;
    _Float16 w0 = Wl[k0 * dout + nc];
    _Float16 w1 = Wl[k1 * dout + nc];
    b[2 * v]     = (nok && k < din)     ? w0 : (_Float16)0.0f;
    b[2 * v + 1] = (nok && k + 1 < din) ? w1 : (_Float16)0.0f;
  }
  return b;
}

// Branchless edge -> (src, dst) resolution (self loops appended after NE)
__device__ __forceinline__ void edge_sd(const int* __restrict__ ei, long e,
                                        int& s, int& d) {
  long ec = (e < NE) ? e : (long)(NE - 1);   // clamped, always in-bounds
  int sld = ei[ec];
  int dld = ei[NE + ec];
  int self = (int)(e - NE);
  bool real = (e < NE);
  s = real ? sld : self;
  d = real ? dld : self;
}

// Prefetch the next grid-stride tile of the streaming operands.
__device__ __forceinline__ void prefetch_edge_tile(const float* __restrict__ ea,
                                                   const int* __restrict__ ei,
                                                   long e_next) {
  long ep = (e_next < EA_TOT) ? e_next : (long)(EA_TOT - 1);
  __builtin_prefetch(ea + ep * (long)LATD, 0, 3);   // -> global_prefetch_b8
  long ec = (ep < NE) ? ep : (long)(NE - 1);
  __builtin_prefetch(ei + ec, 0, 3);
  __builtin_prefetch(ei + NE + ec, 0, 3);
}

// ---------------------------------------------------------------------------
// K1: y = X @ W (DINx64) + b, accumulate per-feature sum / sumsq only.
// ---------------------------------------------------------------------------
template <int DIN, bool GATHER>
__global__ __launch_bounds__(256) void lin64_stats_kernel(
    const float* __restrict__ X, const float* __restrict__ h,
    const float* __restrict__ ea, const int* __restrict__ ei,
    const float* __restrict__ W, const float* __restrict__ bias,
    float* __restrict__ g_sum, float* __restrict__ g_sq, long nrows) {
  constexpr int KCH = (DIN + 31) / 32;
  __shared__ _Float16 Wl[DIN * 64];
  __shared__ float s_sum[64], s_sq[64];
  int tid = threadIdx.x;
  for (int i = tid; i < DIN * 64; i += blockDim.x) Wl[i] = (_Float16)W[i];
  if (tid < 64) { s_sum[tid] = 0.f; s_sq[tid] = 0.f; }
  __syncthreads();

  int lane = tid & 31, wave = tid >> 5;
  long ntiles = nrows >> 4;
  long tstride = (long)gridDim.x * 8;

  // loop-invariant: B fragments and bias columns in registers
  v16h bf[KCH][4];
#pragma unroll
  for (int kc = 0; kc < KCH; ++kc)
#pragma unroll
    for (int nt = 0; nt < 4; ++nt) bf[kc][nt] = b_frag(Wl, DIN, 64, kc, nt, lane);
  float bcol[4];
#pragma unroll
  for (int nt = 0; nt < 4; ++nt) bcol[nt] = bias[nt * 16 + (lane & 15)];

  float accS[4] = {0.f, 0.f, 0.f, 0.f}, accQ[4] = {0.f, 0.f, 0.f, 0.f};

  for (long t = (long)blockIdx.x * 8 + wave; t < ntiles; t += tstride) {
    long row0 = t << 4;
    v16h a[KCH];
    if constexpr (GATHER) {
      long e = row0 + (lane & 15);
      prefetch_edge_tile(ea, ei, e + (tstride << 4));
      int s, d;
      edge_sd(ei, e, s, d);
      a[0] = a_frag_msg(h, ea, s, d, e, 0, lane);
      a[1] = a_frag_msg(h, ea, s, d, e, 1, lane);
      a[2] = a_frag_msg(h, ea, s, d, e, 2, lane);
    } else {
      long rnext = ((t + tstride) << 4) + (lane & 15);
      long rp = (rnext < nrows) ? rnext : (nrows - 1);
      __builtin_prefetch(X + rp * (long)DIN, 0, 3);
#pragma unroll
      for (int kc = 0; kc < KCH; ++kc) a[kc] = a_frag_rows<DIN>(X, row0, kc, lane);
    }
#pragma unroll
    for (int nt = 0; nt < 4; ++nt) {
      v8f c;
#pragma unroll
      for (int r = 0; r < 8; ++r) c[r] = bcol[nt];
#pragma unroll
      for (int kc = 0; kc < KCH; ++kc) c = wmma16(a[kc], bf[kc][nt], c);
#pragma unroll
      for (int r = 0; r < 8; ++r) { accS[nt] += c[r]; accQ[nt] += c[r] * c[r]; }
    }
  }
#pragma unroll
  for (int nt = 0; nt < 4; ++nt) {
    atomicAdd(&s_sum[nt * 16 + (lane & 15)], accS[nt]);
    atomicAdd(&s_sq[nt * 16 + (lane & 15)], accQ[nt]);
  }
  __syncthreads();
  if (tid < 64) {
    atomicAdd(&g_sum[tid], s_sum[tid]);
    atomicAdd(&g_sq[tid], s_sq[tid]);
  }
}

// ---------------------------------------------------------------------------
// K2: recompute y1 = X@W1+b1, BN(scale/shift)+ReLU, LDS restage,
// out = act @ W2 (64xDOUT) + b2.  SCATTER: atomic-add rows into aggr[dst].
// STATS_OUT: accumulate stats of out.
// ---------------------------------------------------------------------------
template <int DIN, int DOUT, bool GATHER, bool STATS_OUT, bool SCATTER>
__global__ __launch_bounds__(256) void lin64_bnrelu_lin_kernel(
    const float* __restrict__ X, const float* __restrict__ h,
    const float* __restrict__ ea, const int* __restrict__ ei,
    const float* __restrict__ W1, const float* __restrict__ b1,
    const float* __restrict__ scale, const float* __restrict__ shift,
    const float* __restrict__ W2, const float* __restrict__ b2,
    float* __restrict__ out, float* __restrict__ aggr,
    float* __restrict__ g_sum, float* __restrict__ g_sq, long nrows) {
  constexpr int KCH = (DIN + 31) / 32;
  constexpr int NT2 = (DOUT + 15) / 16;
  __shared__ _Float16 Wl1[DIN * 64];
  __shared__ _Float16 Wl2[64 * DOUT];
  __shared__ _Float16 stg[8][16][72];
  __shared__ float s_scale[64], s_shift[64];
  __shared__ float s_sum[64], s_sq[64];
  __shared__ int lds_dst[8][16];

  int tid = threadIdx.x;
  for (int i = tid; i < DIN * 64; i += blockDim.x) Wl1[i] = (_Float16)W1[i];
  for (int i = tid; i < 64 * DOUT; i += blockDim.x) Wl2[i] = (_Float16)W2[i];
  if (tid < 64) {
    s_scale[tid] = scale[tid];
    s_shift[tid] = shift[tid];
    s_sum[tid] = 0.f; s_sq[tid] = 0.f;
  }
  __syncthreads();

  int lane = tid & 31, wave = tid >> 5;
  int col = lane & 15;
  int rhi = (lane & 16) ? 8 : 0;
  long ntiles = nrows >> 4;
  long tstride = (long)gridDim.x * 8;

  // loop-invariant fragments / constants in registers
  v16h bf1[KCH][4];
#pragma unroll
  for (int kc = 0; kc < KCH; ++kc)
#pragma unroll
    for (int nt = 0; nt < 4; ++nt) bf1[kc][nt] = b_frag(Wl1, DIN, 64, kc, nt, lane);
  v16h bf2[2][NT2];
#pragma unroll
  for (int kc = 0; kc < 2; ++kc)
#pragma unroll
    for (int nt = 0; nt < NT2; ++nt) bf2[kc][nt] = b_frag(Wl2, 64, DOUT, kc, nt, lane);

  float b1col[4], sccol[4], shcol[4];
#pragma unroll
  for (int nt = 0; nt < 4; ++nt) {
    int cc = nt * 16 + col;
    b1col[nt] = b1[cc];
    sccol[nt] = s_scale[cc];
    shcol[nt] = s_shift[cc];
  }
  float b2col[NT2];
#pragma unroll
  for (int nt = 0; nt < NT2; ++nt) {
    int n = nt * 16 + col;
    b2col[nt] = (n < DOUT) ? b2[n] : 0.f;
  }
  float accS[NT2], accQ[NT2];
#pragma unroll
  for (int nt = 0; nt < NT2; ++nt) { accS[nt] = 0.f; accQ[nt] = 0.f; }

  for (long t = (long)blockIdx.x * 8 + wave; t < ntiles; t += tstride) {
    long row0 = t << 4;
    v16h a[KCH];
    if constexpr (GATHER) {
      long e = row0 + (lane & 15);
      prefetch_edge_tile(ea, ei, e + (tstride << 4));
      int s, d;
      edge_sd(ei, e, s, d);
      if constexpr (SCATTER) { if (lane < 16) lds_dst[wave][lane] = d; }
      a[0] = a_frag_msg(h, ea, s, d, e, 0, lane);
      a[1] = a_frag_msg(h, ea, s, d, e, 1, lane);
      a[2] = a_frag_msg(h, ea, s, d, e, 2, lane);
    } else {
      long rnext = ((t + tstride) << 4) + (lane & 15);
      long rp = (rnext < nrows) ? rnext : (nrows - 1);
      __builtin_prefetch(X + rp * (long)DIN, 0, 3);
#pragma unroll
      for (int kc = 0; kc < KCH; ++kc) a[kc] = a_frag_rows<DIN>(X, row0, kc, lane);
    }
    // ---- GEMM1 + BN + ReLU -> per-wave LDS stage (C-layout -> A-layout) ----
#pragma unroll
    for (int nt = 0; nt < 4; ++nt) {
      v8f c;
#pragma unroll
      for (int r = 0; r < 8; ++r) c[r] = b1col[nt];
#pragma unroll
      for (int kc = 0; kc < KCH; ++kc) c = wmma16(a[kc], bf1[kc][nt], c);
      int cc = nt * 16 + col;
#pragma unroll
      for (int r = 0; r < 8; ++r) {
        float v = fmaxf(c[r] * sccol[nt] + shcol[nt], 0.f);   // BN + ReLU
        stg[wave][r + rhi][cc] = (_Float16)v;
      }
    }
    // ---- GEMM2: (16x64) @ (64xDOUT) ---- (DS ops in-order within a wave)
    v16h a2_0 = a_frag_stage(&stg[wave][0][0], 0, lane);
    v16h a2_1 = a_frag_stage(&stg[wave][0][0], 1, lane);
#pragma unroll
    for (int nt = 0; nt < NT2; ++nt) {
      v8f c;
#pragma unroll
      for (int r = 0; r < 8; ++r) c[r] = b2col[nt];
      c = wmma16(a2_0, bf2[0][nt], c);
      c = wmma16(a2_1, bf2[1][nt], c);
      int n = nt * 16 + col;
      if constexpr (SCATTER) {
#pragma unroll
        for (int r = 0; r < 8; ++r) {
          int dd = lds_dst[wave][r + rhi];
          atomicAdd(&aggr[(long)dd * 64 + n], c[r]);
        }
      } else {
        if (n < DOUT) {
#pragma unroll
          for (int r = 0; r < 8; ++r)
            out[(row0 + r + rhi) * (long)DOUT + n] = c[r];
        }
        if constexpr (STATS_OUT) {
#pragma unroll
          for (int r = 0; r < 8; ++r) { accS[nt] += c[r]; accQ[nt] += c[r] * c[r]; }
        }
      }
    }
  }
  if constexpr (STATS_OUT) {
#pragma unroll
    for (int nt = 0; nt < NT2; ++nt) {
      atomicAdd(&s_sum[nt * 16 + (lane & 15)], accS[nt]);
      atomicAdd(&s_sq[nt * 16 + (lane & 15)], accQ[nt]);
    }
    __syncthreads();
    if (tid < 64) {
      atomicAdd(&g_sum[tid], s_sum[tid]);
      atomicAdd(&g_sq[tid], s_sq[tid]);
    }
  }
}

// ---------------------------------------------------------------------------
// K3: out = relu(BN(Y)) @ W (64xDOUT) + b    (Y is rows x 64 f32)
// ---------------------------------------------------------------------------
template <int DOUT>
__global__ __launch_bounds__(256) void bnrelu_lin_kernel(
    const float* __restrict__ Y, const float* __restrict__ scale,
    const float* __restrict__ shift, const float* __restrict__ W,
    const float* __restrict__ bias, float* __restrict__ out,
    float* __restrict__ out2, long nrows) {
  constexpr int NT2 = (DOUT + 15) / 16;
  __shared__ _Float16 Wl[64 * DOUT];
  __shared__ float s_scale[64], s_shift[64];
  int tid = threadIdx.x;
  for (int i = tid; i < 64 * DOUT; i += blockDim.x) Wl[i] = (_Float16)W[i];
  if (tid < 64) { s_scale[tid] = scale[tid]; s_shift[tid] = shift[tid]; }
  __syncthreads();

  int lane = tid & 31, wave = tid >> 5;
  int col = lane & 15;
  int rhi = (lane & 16) ? 8 : 0;
  long ntiles = nrows >> 4;
  long tstride = (long)gridDim.x * 8;

  v16h bfw[2][NT2];
#pragma unroll
  for (int kc = 0; kc < 2; ++kc)
#pragma unroll
    for (int nt = 0; nt < NT2; ++nt) bfw[kc][nt] = b_frag(Wl, 64, DOUT, kc, nt, lane);
  float bcol[NT2];
#pragma unroll
  for (int nt = 0; nt < NT2; ++nt) {
    int n = nt * 16 + col;
    bcol[nt] = (n < DOUT) ? bias[n] : 0.f;
  }
  // per-lane BN constants for the 32 K-values this lane touches
  float scv[2][16], shv[2][16];
#pragma unroll
  for (int kc = 0; kc < 2; ++kc)
#pragma unroll
    for (int v = 0; v < 8; ++v) {
      int k = kc * 32 + kmap(v, lane);
      scv[kc][2 * v] = s_scale[k];         shv[kc][2 * v] = s_shift[k];
      scv[kc][2 * v + 1] = s_scale[k + 1]; shv[kc][2 * v + 1] = s_shift[k + 1];
    }

  for (long t = (long)blockIdx.x * 8 + wave; t < ntiles; t += tstride) {
    long row0 = t << 4;
    const float* p = Y + (row0 + (lane & 15)) * 64;
    {
      long rnext = ((t + tstride) << 4) + (lane & 15);
      long rp = (rnext < nrows) ? rnext : (nrows - 1);
      __builtin_prefetch(Y + rp * 64, 0, 3);
    }
    v16h a[2];
#pragma unroll
    for (int kc = 0; kc < 2; ++kc) {
#pragma unroll
      for (int v = 0; v < 8; ++v) {
        int k = kc * 32 + kmap(v, lane);
        float2 f = *(const float2*)(p + k);
        a[kc][2 * v]     = (_Float16)fmaxf(f.x * scv[kc][2 * v] + shv[kc][2 * v], 0.f);
        a[kc][2 * v + 1] = (_Float16)fmaxf(f.y * scv[kc][2 * v + 1] + shv[kc][2 * v + 1], 0.f);
      }
    }
#pragma unroll
    for (int nt = 0; nt < NT2; ++nt) {
      v8f c;
#pragma unroll
      for (int r = 0; r < 8; ++r) c[r] = bcol[nt];
      c = wmma16(a[0], bfw[0][nt], c);
      c = wmma16(a[1], bfw[1][nt], c);
      int n = nt * 16 + col;
      if (n < DOUT) {
#pragma unroll
        for (int r = 0; r < 8; ++r) {
          float v = c[r];
          long idx = (row0 + r + rhi) * (long)DOUT + n;
          out[idx] = v;
          if (out2) out2[idx] = v;
        }
      }
    }
  }
}

// ---------------------------------------------------------------------------
// small helper kernels
// ---------------------------------------------------------------------------
__global__ void bn_finalize_kernel(const float* g_sum, const float* g_sq,
                                   const float* gamma, const float* beta,
                                   float* scale, float* shift, float inv_cnt) {
  int i = threadIdx.x;  // 64
  float mu = g_sum[i] * inv_cnt;
  float var = g_sq[i] * inv_cnt - mu * mu;
  float sc = gamma[i] * rsqrtf(var + EPSBN);
  scale[i] = sc;
  shift[i] = beta[i] - mu * sc;
}

__global__ void zero_kernel(float* p, long n) {
  for (long i = blockIdx.x * (long)blockDim.x + threadIdx.x; i < n;
       i += (long)gridDim.x * blockDim.x) p[i] = 0.f;
}
__global__ void fill_ones_kernel(float* p, long n) {
  for (long i = blockIdx.x * (long)blockDim.x + threadIdx.x; i < n;
       i += (long)gridDim.x * blockDim.x) p[i] = 1.f;
}
__global__ void counts_kernel(const int* __restrict__ ei, float* counts) {
  long e = blockIdx.x * (long)blockDim.x + threadIdx.x;
  if (e >= EA_TOT) return;
  int d = (e < NE) ? ei[NE + e] : (int)(e - NE);
  atomicAdd(&counts[d], 1.f);
}
__global__ void invert_counts_kernel(float* counts) {
  long i = blockIdx.x * (long)blockDim.x + threadIdx.x;
  if (i < NN) counts[i] = 1.f / fmaxf(counts[i], 1.f);
}
// upd_in[n, 0:32] = h[n]; upd_in[n, 32:96] = aggr[n] * inv_count[n]
__global__ void build_upd_in_kernel(const float* __restrict__ h,
                                    const float* __restrict__ aggr,
                                    const float* __restrict__ invc,
                                    float* __restrict__ upd_in) {
  long i = blockIdx.x * (long)blockDim.x + threadIdx.x;
  if (i >= (long)NN * 96) return;
  long n = i / 96; int j = (int)(i % 96);
  upd_in[i] = (j < 32) ? h[n * 32 + j] : aggr[n * 64 + (j - 32)] * invc[n];
}
__global__ void residual_kernel(const float* a, const float* b, float* o, long n) {
  for (long i = blockIdx.x * (long)blockDim.x + threadIdx.x; i < n;
       i += (long)gridDim.x * blockDim.x) o[i] = a[i] + b[i];
}

// ---------------------------------------------------------------------------
// host launch
// ---------------------------------------------------------------------------
static inline int tile_grid(long nrows) {
  long b = ((nrows >> 4) + 7) / 8;
  return (int)(b < 4096 ? b : 4096);
}
static inline int ew_grid(long n) {
  long b = (n + 255) / 256;
  return (int)(b < 16384 ? b : 16384);
}

extern "C" void kernel_launch(void* const* d_in, const int* in_sizes, int n_in,
                              void* d_out, int out_size, void* d_ws, size_t ws_size,
                              hipStream_t stream) {
  (void)in_sizes; (void)n_in; (void)out_size; (void)ws_size;
  auto F = [&](int i) { return (const float*)d_in[i]; };
  const float* x     = F(0);                     // [NN,3]
  const int*   ei    = (const int*)d_in[1];      // [2,NE]
  const float* eattr = F(2);                     // [NE,4]
  // params flattened in JAX pytree order (dict keys sorted alphabetically):
  // 3..12 decoder | 13..18 edge_encoder | 19..54 mp_layers | 55..64 node_encoder
  // each MLP: final.W, final.b, then per hidden: beta, gamma, lin.W, lin.b

  float* ws = (float*)d_ws;
  size_t off = 0;
  auto A = [&](size_t n) { float* p = ws + off; off += n; return p; };
  float* node_latent = A((size_t)NN * 32);
  float* h           = A((size_t)NN * 32);
  float* ea          = A((size_t)EA_TOT * 32);
  float* ybuf        = A((size_t)NN * 64);
  float* aggr        = A((size_t)NN * 64);
  float* upd_in      = A((size_t)NN * 96);
  float* upd_out     = A((size_t)NN * 32);
  float* counts      = A(NN);
  float* g_sum  = A(64); float* g_sq  = A(64);
  float* g_sum2 = A(64); float* g_sq2 = A(64);
  float* bnA_sc = A(64); float* bnA_sh = A(64);
  float* bnB_sc = A(64); float* bnB_sh = A(64);

  dim3 B(256);
  int gN  = tile_grid(NN);
  int gE  = tile_grid(NE);
  int gEa = tile_grid(EA_TOT);

  // ===================== node encoder: 3 ->64 BN ReLU ->64 BN ReLU ->32 ====
  zero_kernel<<<1, 128, 0, stream>>>(g_sum, 128);
  lin64_stats_kernel<3, false><<<gN, B, 0, stream>>>(
      x, nullptr, nullptr, nullptr, F(59), F(60), g_sum, g_sq, NN);
  bn_finalize_kernel<<<1, 64, 0, stream>>>(g_sum, g_sq, F(58), F(57),
                                           bnA_sc, bnA_sh, 1.f / NN);
  zero_kernel<<<1, 128, 0, stream>>>(g_sum2, 128);
  lin64_bnrelu_lin_kernel<3, 64, false, true, false><<<gN, B, 0, stream>>>(
      x, nullptr, nullptr, nullptr, F(59), F(60), bnA_sc, bnA_sh, F(63), F(64),
      ybuf, nullptr, g_sum2, g_sq2, NN);
  bn_finalize_kernel<<<1, 64, 0, stream>>>(g_sum2, g_sq2, F(62), F(61),
                                           bnB_sc, bnB_sh, 1.f / NN);
  bnrelu_lin_kernel<32><<<gN, B, 0, stream>>>(ybuf, bnB_sc, bnB_sh, F(55), F(56),
                                              node_latent, h, NN);

  // ===================== edge encoder: 4 ->64 BN ReLU ->32 =================
  zero_kernel<<<1, 128, 0, stream>>>(g_sum, 128);
  lin64_stats_kernel<4, false><<<gE, B, 0, stream>>>(
      eattr, nullptr, nullptr, nullptr, F(17), F(18), g_sum, g_sq, NE);
  bn_finalize_kernel<<<1, 64, 0, stream>>>(g_sum, g_sq, F(16), F(15),
                                           bnA_sc, bnA_sh, 1.f / NE);
  lin64_bnrelu_lin_kernel<4, 32, false, false, false><<<gE, B, 0, stream>>>(
      eattr, nullptr, nullptr, nullptr, F(17), F(18), bnA_sc, bnA_sh, F(13), F(14),
      ea, nullptr, nullptr, nullptr, NE);
  fill_ones_kernel<<<ew_grid((long)NN * 32), B, 0, stream>>>(
      ea + (size_t)NE * 32, (long)NN * 32);   // self-loop edge_attr = 1.0

  // ===================== in-degree (incl. self loops) ======================
  zero_kernel<<<ew_grid(NN), B, 0, stream>>>(counts, NN);
  counts_kernel<<<ew_grid(EA_TOT), B, 0, stream>>>(ei, counts);
  invert_counts_kernel<<<ew_grid(NN), B, 0, stream>>>(counts);

  // ===================== 3 message-passing layers ==========================
  for (int i = 0; i < 3; ++i) {
    int mb = 19 + i * 12;  // message: mb..mb+5, update: mb+6..mb+11
    zero_kernel<<<1, 128, 0, stream>>>(g_sum, 128);
    lin64_stats_kernel<96, true><<<gEa, B, 0, stream>>>(
        nullptr, h, ea, ei, F(mb + 4), F(mb + 5), g_sum, g_sq, EA_TOT);
    bn_finalize_kernel<<<1, 64, 0, stream>>>(g_sum, g_sq, F(mb + 3), F(mb + 2),
                                             bnA_sc, bnA_sh, 1.f / EA_TOT);
    zero_kernel<<<ew_grid((long)NN * 64), B, 0, stream>>>(aggr, (long)NN * 64);
    lin64_bnrelu_lin_kernel<96, 64, true, false, true><<<gEa, B, 0, stream>>>(
        nullptr, h, ea, ei, F(mb + 4), F(mb + 5), bnA_sc, bnA_sh,
        F(mb), F(mb + 1), nullptr, aggr, nullptr, nullptr, EA_TOT);
    build_upd_in_kernel<<<ew_grid((long)NN * 96), B, 0, stream>>>(h, aggr, counts,
                                                                  upd_in);
    zero_kernel<<<1, 128, 0, stream>>>(g_sum, 128);
    lin64_stats_kernel<96, false><<<gN, B, 0, stream>>>(
        upd_in, nullptr, nullptr, nullptr, F(mb + 10), F(mb + 11), g_sum, g_sq, NN);
    bn_finalize_kernel<<<1, 64, 0, stream>>>(g_sum, g_sq, F(mb + 9), F(mb + 8),
                                             bnA_sc, bnA_sh, 1.f / NN);
    lin64_bnrelu_lin_kernel<96, 32, false, false, false><<<gN, B, 0, stream>>>(
        upd_in, nullptr, nullptr, nullptr, F(mb + 10), F(mb + 11), bnA_sc, bnA_sh,
        F(mb + 6), F(mb + 7), upd_out, nullptr, nullptr, nullptr, NN);
    residual_kernel<<<ew_grid((long)NN * 32), B, 0, stream>>>(upd_out, node_latent,
                                                              h, (long)NN * 32);
  }

  // ===================== decoder: 32 ->64 BN ReLU ->64 BN ReLU ->4 =========
  zero_kernel<<<1, 128, 0, stream>>>(g_sum, 128);
  lin64_stats_kernel<32, false><<<gN, B, 0, stream>>>(
      h, nullptr, nullptr, nullptr, F(7), F(8), g_sum, g_sq, NN);
  bn_finalize_kernel<<<1, 64, 0, stream>>>(g_sum, g_sq, F(6), F(5),
                                           bnA_sc, bnA_sh, 1.f / NN);
  zero_kernel<<<1, 128, 0, stream>>>(g_sum2, 128);
  lin64_bnrelu_lin_kernel<32, 64, false, true, false><<<gN, B, 0, stream>>>(
      h, nullptr, nullptr, nullptr, F(7), F(8), bnA_sc, bnA_sh, F(11), F(12),
      ybuf, nullptr, g_sum2, g_sq2, NN);
  bn_finalize_kernel<<<1, 64, 0, stream>>>(g_sum2, g_sq2, F(10), F(9),
                                           bnB_sc, bnB_sh, 1.f / NN);
  bnrelu_lin_kernel<4><<<gN, B, 0, stream>>>(ybuf, bnB_sc, bnB_sh, F(3), F(4),
                                             (float*)d_out, nullptr, NN);
}
